// NeuralAdditiveModel_81930796138712
// MI455X (gfx1250) — compile-verified, matrix operand order checked by reading, not run on hardware
//
#include <hip/hip_runtime.h>

typedef __attribute__((ext_vector_type(16))) _Float16 v16h;
typedef __attribute__((ext_vector_type(8)))  _Float16 v8h;
typedef __attribute__((ext_vector_type(8)))  float    v8f;

#define B_   8192
#define F_   256
#define H1_  64
#define H2_  32
#define C_   10
#define FPC  8                 // features staged per LDS chunk
#define NCHUNK (F_ / FPC)

// WMMA 16-bit operand K index for half-slot i (0..15) in lane-group g (0/1):
//   K = i + 8g        for i < 8
//   K = i + 8 + 8g    for i >= 8
__device__ __forceinline__ int kmap(int i, int g) {
    return (i < 8) ? (i + 8 * g) : (i + 8 + 8 * g);
}

__launch_bounds__(256, 1)
__global__ void nam_wmma_kernel(const float* __restrict__ x,
                                const float* __restrict__ W1,
                                const float* __restrict__ b1,
                                const float* __restrict__ W2,
                                const float* __restrict__ b2,
                                const float* __restrict__ Wout,
                                const float* __restrict__ bias,
                                float* __restrict__ out)
{
    // ---- LDS staging buffers (35 KB total) ----
    __shared__ __align__(32) _Float16 sW2[FPC][2][2][32][16]; // [f8][kblk][nblk][lane][i] 16KB
    __shared__ __align__(32) _Float16 sWo[FPC][32][16];       // [f8][lane][i]              8KB
    __shared__ __align__(32) _Float16 sW1[FPC][2][2][16];     // [f8][kblk][g][i] f16       1KB
    __shared__ __align__(32) _Float16 sB1[FPC][2][2][16];     //                            1KB
    __shared__ float sB2[FPC][32];                            //                            1KB
    __shared__ __align__(16) _Float16 sH2[8][16][32];         // per-wave h2 transpose      8KB

    const int tid  = threadIdx.x;
    const int lane = tid & 31;
    const int wave = tid >> 5;
    const int g    = lane >> 4;   // lane group (K-half select)
    const int m    = lane & 15;   // row (A) / column N (B,C,D)
    const int b0   = (blockIdx.x * 8 + wave) * 16;   // this wave's batch tile

    // persistent 16x16 f32 C-tile, classifier bias pre-folded into the accumulator
    const float bv = (m < C_) ? bias[m] : 0.0f;
    v8f accOut;
    #pragma unroll
    for (int r = 0; r < 8; ++r) accOut[r] = bv;

    for (int chunk = 0; chunk < NCHUNK; ++chunk) {
        const int fbase = chunk * FPC;
        __syncthreads();   // previous chunk's compute done before overwriting LDS

        // ---- cooperative gather+convert of packed weights into LDS ----
        for (int idx = tid; idx < FPC * 2048; idx += 256) {   // W2 -> B layout, f16
            int i  = idx & 15;
            int ln = (idx >> 4) & 31;
            int nb = (idx >> 9) & 1;
            int kb = (idx >> 10) & 1;
            int f8 = idx >> 11;
            int K  = kmap(i, ln >> 4) + 32 * kb;
            int o  = (ln & 15) + 16 * nb;
            sW2[f8][kb][nb][ln][i] =
                (_Float16)W2[((fbase + f8) * H1_ + K) * H2_ + o];
        }
        for (int idx = tid; idx < FPC * 512; idx += 256) {    // Wout -> B layout, pad N to 16
            int i  = idx & 15;
            int ln = (idx >> 4) & 31;
            int f8 = idx >> 9;
            int n  = ln & 15;
            int K  = kmap(i, ln >> 4);
            float v = (n < C_) ? Wout[((fbase + f8) * H2_ + K) * C_ + n] : 0.0f;
            sWo[f8][ln][i] = (_Float16)v;
        }
        for (int idx = tid; idx < FPC * 64; idx += 256) {     // W1 / b1 f16 in A-reg K order
            int i  = idx & 15;
            int gg = (idx >> 4) & 1;
            int kb = (idx >> 5) & 1;
            int f8 = idx >> 6;
            int K  = kmap(i, gg) + 32 * kb;
            sW1[f8][kb][gg][i] = (_Float16)W1[(fbase + f8) * H1_ + K];
            sB1[f8][kb][gg][i] = (_Float16)b1[(fbase + f8) * H1_ + K];
        }
        for (int idx = tid; idx < FPC * 32; idx += 256) {     // b2 (f32)
            sB2[idx >> 5][idx & 31] = b2[(fbase + (idx >> 5)) * H2_ + (idx & 31)];
        }
        __syncthreads();

        // ---- per-feature fused MLP on this wave's 16-row tile ----
        for (int f8 = 0; f8 < FPC; ++f8) {
            const int f = fbase + f8;
            const _Float16 xh = (_Float16)x[(b0 + m) * F_ + f];

            // Layer 1 in packed f16: a = relu(x * W1 + b1) -> v_pk_fma_f16 / v_pk_max
            v16h xs;
            #pragma unroll
            for (int i = 0; i < 16; ++i) xs[i] = xh;
            const v16h w10 = *(const v16h*)&sW1[f8][0][g][0];
            const v16h w11 = *(const v16h*)&sW1[f8][1][g][0];
            const v16h c10 = *(const v16h*)&sB1[f8][0][g][0];
            const v16h c11 = *(const v16h*)&sB1[f8][1][g][0];
            const v16h zero = {};
            v16h a0 = __builtin_elementwise_max((v16h)(xs * w10 + c10), zero);
            v16h a1 = __builtin_elementwise_max((v16h)(xs * w11 + c11), zero);

            // Layer 2: [16,64] @ [64,32], b2 bias pre-loaded into the C operand
            const v16h b00 = *(const v16h*)&sW2[f8][0][0][lane][0];
            const v16h b10 = *(const v16h*)&sW2[f8][1][0][lane][0];
            const v16h b01 = *(const v16h*)&sW2[f8][0][1][lane][0];
            const v16h b11 = *(const v16h*)&sW2[f8][1][1][lane][0];
            const float bz0 = sB2[f8][m];
            const float bz1 = sB2[f8][m + 16];
            v8f acc0, acc1;
            #pragma unroll
            for (int r = 0; r < 8; ++r) { acc0[r] = bz0; acc1[r] = bz1; }
            acc0 = __builtin_amdgcn_wmma_f32_16x16x32_f16(false, a0, false, b00,
                                                          (short)0, acc0, false, false);
            acc0 = __builtin_amdgcn_wmma_f32_16x16x32_f16(false, a1, false, b10,
                                                          (short)0, acc0, false, false);
            acc1 = __builtin_amdgcn_wmma_f32_16x16x32_f16(false, a0, false, b01,
                                                          (short)0, acc1, false, false);
            acc1 = __builtin_amdgcn_wmma_f32_16x16x32_f16(false, a1, false, b11,
                                                          (short)0, acc1, false, false);

            // packed convert (v_cvt_pk_f16_f32) then packed ReLU (v_pk_max_num_f16);
            // relu after cvt == cvt after relu (monotone, 0 -> 0)
            v16h hv;
            #pragma unroll
            for (int r = 0; r < 8; ++r) {
                hv[2 * r]     = (_Float16)acc0[r];
                hv[2 * r + 1] = (_Float16)acc1[r];
            }
            hv = __builtin_elementwise_max(hv, zero);

            // park h2 row-major in LDS (C layout -> canonical h2[M][o])
            #pragma unroll
            for (int r = 0; r < 8; ++r) {
                const int M = r + 8 * g;
                sH2[wave][M][m]      = hv[2 * r];       // o = m      (nblk 0)
                sH2[wave][M][m + 16] = hv[2 * r + 1];   // o = m + 16 (nblk 1)
            }
            asm volatile("s_wait_dscnt 0" ::: "memory");  // DS RAW fence (same wave)

            // reload h2 as A-tile (16x32, K order per lane group) -- lands in slot order
            const _Float16* row = &sH2[wave][m][0];
            const v8h lo = *(const v8h*)&row[8 * g];
            const v8h hi = *(const v8h*)&row[16 + 8 * g];
            v16h aH;
            #pragma unroll
            for (int j = 0; j < 8; ++j) { aH[j] = lo[j]; aH[j + 8] = hi[j]; }

            // Output layer: [16,32] @ [32,16pad], accumulate into persistent C-tile
            const v16h bo = *(const v16h*)&sWo[f8][lane][0];
            accOut = __builtin_amdgcn_wmma_f32_16x16x32_f16(false, aH, false, bo,
                                                            (short)0, accOut, false, false);
        }
    }

    // ---- epilogue: pure copy-out (bias already folded into accOut init) ----
    if (m < C_) {
        #pragma unroll
        for (int r = 0; r < 8; ++r) {
            out[(b0 + r + 8 * g) * C_ + m] = accOut[r];
        }
    }
}

extern "C" void kernel_launch(void* const* d_in, const int* in_sizes, int n_in,
                              void* d_out, int out_size, void* d_ws, size_t ws_size,
                              hipStream_t stream) {
    (void)in_sizes; (void)n_in; (void)d_ws; (void)ws_size; (void)out_size;
    const float* x    = (const float*)d_in[0];
    const float* W1   = (const float*)d_in[1];
    const float* b1   = (const float*)d_in[2];
    const float* W2   = (const float*)d_in[3];
    const float* b2   = (const float*)d_in[4];
    const float* Wout = (const float*)d_in[5];
    const float* bias = (const float*)d_in[6];
    float* out = (float*)d_out;

    // 512 batch tiles of 16 rows, 8 wave32-tiles per 256-thread block -> 64 blocks
    nam_wmma_kernel<<<dim3(B_ / (16 * 8)), dim3(256), 0, stream>>>(
        x, W1, b1, W2, b2, Wout, bias, out);
}
